// SpatialContextEncoder_1726576854241
// MI455X (gfx1250) — matrix operand order repaired and checked by reading, not compiled, not run on hardware
//
#include <hip/hip_runtime.h>

// ---------------------------------------------------------------------------
// SpatialContextEncoder for MI455X (gfx1250, wave32, bf16 WMMA, async-to-LDS)
// Activations kept in NHWC ([spatial, channel]) so both WMMA fragments are
// K-contiguous: A = act[s][k], B = W[o][k] (= B-matrix K x N transposed view).
// GEMM: D[65536, 512] = act[65536, K] x W[512, K]^T, f32 accumulate.
// ---------------------------------------------------------------------------

typedef __attribute__((ext_vector_type(16))) __bf16 v16bf;
typedef __attribute__((ext_vector_type(8)))  float  v8f;

#define HW 4096                 // 64*64 spatial per image
#define NB 16                   // batch
#define NS ((long)NB * HW)      // 65536 total spatial rows

#define TM 128                  // block tile M (spatial)
#define TN 128                  // block tile N (out channels)
#define TK 32                   // K chunk = one WMMA K

__device__ __forceinline__ unsigned short f2bf(float f) {
  unsigned u = __float_as_uint(f);
  unsigned r = u + 0x7FFFu + ((u >> 16) & 1u);   // round-to-nearest-even
  return (unsigned short)(r >> 16);
}
__device__ __forceinline__ float bf2f(unsigned short h) {
  return __uint_as_float(((unsigned)h) << 16);
}

// async copy 16B global -> LDS (ASYNCcnt-tracked, ISA 10 / 15.18.3 op 98)
#define ASYNC_B128(ldsAddr, gPtr)                                             \
  {                                                                           \
    unsigned long long __g = (unsigned long long)(size_t)(gPtr);              \
    asm volatile("global_load_async_to_lds_b128 %0, %1, off"                  \
                 :: "v"(ldsAddr), "v"(__g) : "memory");                       \
  }
#define WAIT_ASYNC_4 asm volatile("s_wait_asynccnt 0x4" ::: "memory")

// ----------------- NHWC bf16 WMMA GEMM with async double buffer ------------
// ACT: bf16 [NS, K]   (row-major, K mult of 32)
// W  : bf16 [512, K]  (row-major)
// D  : f32  [NS, 512]
__global__ __launch_bounds__(256) void gemm_nhwc_wmma(
    const unsigned short* __restrict__ ACT,
    const unsigned short* __restrict__ W,
    float* __restrict__ D,
    int K)
{
  __shared__ unsigned short As[2][TM * TK];   // 2 x 8 KB
  __shared__ unsigned short Ws[2][TN * TK];   // 2 x 8 KB

  const int m0   = blockIdx.y * TM;
  const int n0   = blockIdx.x * TN;
  const int tid  = threadIdx.x;
  const int lane = tid & 31;
  const int wave = tid >> 5;            // 0..7
  const int wm   = (wave & 3) * 32;     // 4 waves along M, 32 rows each
  const int wn   = (wave >> 2) * 64;    // 2 waves along N, 64 cols each

  // cooperative tile staging: 2 threads per 32-elem row, 16 elems each
  const int trow = tid >> 1;            // 0..127
  const int tcol = (tid & 1) * 16;      // 0 or 16

  const int niter = K / TK;

  v8f acc[2][4] = {};

  // incremental global pointers (advance TK elements per issued tile)
  const unsigned short* ga = ACT + (long)(m0 + trow) * K + tcol;
  const unsigned short* gw = W   + (long)(n0 + trow) * K + tcol;
  // LDS double-buffer byte addresses (flat addr[31:0] == LDS offset)
  const unsigned la0 = (unsigned)(size_t)&As[0][trow * TK + tcol];
  const unsigned la1 = (unsigned)(size_t)&As[1][trow * TK + tcol];
  const unsigned lw0 = (unsigned)(size_t)&Ws[0][trow * TK + tcol];
  const unsigned lw1 = (unsigned)(size_t)&Ws[1][trow * TK + tcol];

  // prologue: tile 0 -> buffer 0
  ASYNC_B128(la0,      ga);
  ASYNC_B128(la0 + 16, ga + 8);
  ASYNC_B128(lw0,      gw);
  ASYNC_B128(lw0 + 16, gw + 8);

  const int half = lane >> 4;     // 0/1: lane half selects K sub-range
  const int ml   = lane & 15;

  union Frag { uint4 q[2]; v16bf v; };

  for (int kt = 0; kt < niter; ++kt) {
    const int cur = kt & 1;
    // branchless next-tile issue: last iteration re-issues the final tile
    // into the spare buffer (never consumed; drained by S_ENDPGM wait-idle).
    const long step = (kt + 1 < niter) ? TK : 0;
    ga += step;  gw += step;
    const unsigned la = cur ? la0 : la1;      // next buffer = cur ^ 1
    const unsigned lw = cur ? lw0 : lw1;
    ASYNC_B128(la,      ga);
    ASYNC_B128(la + 16, ga + 8);
    ASYNC_B128(lw,      gw);
    ASYNC_B128(lw + 16, gw + 8);
    WAIT_ASYNC_4;                 // drain current tile, keep next in flight
    __syncthreads();

    const unsigned short* Ab = &As[cur][0];
    const unsigned short* Wb = &Ws[cur][0];

    // A fragments (16x32 bf16, ISA 7.12.2): per lane two 16B contiguous runs
    Frag af[2];
#pragma unroll
    for (int i = 0; i < 2; ++i) {
      const unsigned short* p = Ab + (wm + i * 16 + ml) * TK + half * 16;
      af[i].q[0] = *(const uint4*)p;          // K = 16*half .. +7
      af[i].q[1] = *(const uint4*)(p + 8);    // K = 16*half+8 .. +7
    }
    // B fragments (32x16 bf16): lane half selects K 0..15 / 16..31, all
    // 16 elements contiguous within the weight row.
    Frag bf[4];
#pragma unroll
    for (int j = 0; j < 4; ++j) {
      const unsigned short* p = Wb + (wn + j * 16 + ml) * TK + half * 16;
      bf[j].q[0] = *(const uint4*)p;
      bf[j].q[1] = *(const uint4*)(p + 8);
    }

#pragma unroll
    for (int i = 0; i < 2; ++i)
#pragma unroll
      for (int j = 0; j < 4; ++j)
        acc[i][j] = __builtin_amdgcn_wmma_f32_16x16x32_bf16(
            false, af[i].v, false, bf[j].v, (short)0, acc[i][j], false, false);

    __syncthreads();              // all reads done before next-iter overwrite
  }

  // store: C/D layout -> M = r + 8*half, N = lane&15 (coalesced along N)
#pragma unroll
  for (int i = 0; i < 2; ++i)
#pragma unroll
    for (int j = 0; j < 4; ++j)
#pragma unroll
      for (int r = 0; r < 8; ++r) {
        int m = m0 + wm + i * 16 + r + half * 8;
        int n = n0 + wn + j * 16 + ml;
        D[(long)m * 512 + n] = acc[i][j][r];
      }
}

// ------------- x: (b, 1024, hw) f32  ->  xb: (b, hw, 1024) bf16 ------------
__global__ __launch_bounds__(256) void transpose_pack_x(
    const float* __restrict__ x, unsigned short* __restrict__ xb) {
  __shared__ float t[32][33];
  const int b  = blockIdx.z;
  const int c0 = blockIdx.y * 32;
  const int s0 = blockIdx.x * 32;
  const int tc = threadIdx.x & 31, tr = threadIdx.x >> 5;   // 8 row-groups
  const float* xp = x + (long)b * 1024 * HW;
  unsigned short* op = xb + (long)b * HW * 1024;
#pragma unroll
  for (int r = tr; r < 32; r += 8)
    t[r][tc] = xp[(long)(c0 + r) * HW + s0 + tc];
  __syncthreads();
#pragma unroll
  for (int r = tr; r < 32; r += 8)
    op[(long)(s0 + r) * 1024 + c0 + tc] = f2bf(t[tc][r]);
}

// --------------------------- weight packing --------------------------------
__global__ void pack_bf16(const float* __restrict__ in,
                          unsigned short* __restrict__ out, long n) {
  long stride = (long)gridDim.x * blockDim.x;
  for (long i = (long)blockIdx.x * blockDim.x + threadIdx.x; i < n; i += stride)
    out[i] = f2bf(in[i]);
}
__global__ void pack_wemb(const float* __restrict__ in,
                          unsigned short* __restrict__ out,
                          int rows, int kin, int kpad) {
  int i = blockIdx.x * blockDim.x + threadIdx.x;
  if (i >= rows * kpad) return;
  int r = i / kpad, c = i - r * kpad;
  out[i] = (c < kin) ? f2bf(in[r * kin + c]) : (unsigned short)0;
}

// ----------- BN stats over columns of D[65536, 512] (two-phase) ------------
__global__ __launch_bounds__(256) void col_stats_partial(
    const float* __restrict__ D, float* __restrict__ psum, float* __restrict__ psq) {
  const int c = blockIdx.y * 256 + threadIdx.x;       // 0..511
  const long r0 = (long)blockIdx.x * (65536 / 32);    // 32 row chunks
  float s = 0.f, s2 = 0.f;
  for (long r = r0; r < r0 + 2048; ++r) {
    float v = D[r * 512 + c];
    s += v;
    s2 = fmaf(v, v, s2);
  }
  psum[blockIdx.x * 512 + c] = s;
  psq [blockIdx.x * 512 + c] = s2;
}
__global__ void col_stats_final(const float* __restrict__ psum,
                                const float* __restrict__ psq,
                                float* __restrict__ mean, float* __restrict__ istd) {
  const int c = blockIdx.x * 256 + threadIdx.x;       // grid 2
  float s = 0.f, s2 = 0.f;
  for (int i = 0; i < 32; ++i) { s += psum[i * 512 + c]; s2 += psq[i * 512 + c]; }
  const float invn = 1.0f / 65536.0f;
  float m = s * invn;
  float var = s2 * invn - m * m;
  mean[c] = m;
  istd[c] = rsqrtf(var + 1e-5f);
}

// ---------- BN + ReLU + bf16 pack, [s][512] -> [s][outStride] --------------
// 8 channels per thread: 2x b128 f32 loads -> 1x b128 bf16 store.
__global__ void bn_relu_pack(const float* __restrict__ D,
                             const float* __restrict__ mean, const float* __restrict__ istd,
                             const float* __restrict__ gamma, const float* __restrict__ beta,
                             unsigned short* __restrict__ out, int outStride) {
  const long total = NS * 64;                         // 8-element groups
  long stride = (long)gridDim.x * blockDim.x;
  for (long g = (long)blockIdx.x * blockDim.x + threadIdx.x; g < total; g += stride) {
    long s  = g >> 6;
    int  c0 = (int)(g & 63) * 8;
    const float* p = D + s * 512 + c0;
    unsigned short r[8];
#pragma unroll
    for (int k = 0; k < 8; ++k) {
      int c = c0 + k;
      float h = (p[k] - mean[c]) * istd[c] * gamma[c] + beta[c];
      r[k] = f2bf(fmaxf(h, 0.f));
    }
    *(uint4*)(out + s * outStride + c0) = *(const uint4*)r;
  }
}

// -------- final BN + transpose back to NCHW f32 output ---------------------
__global__ __launch_bounds__(256) void bn_final_t(
    const float* __restrict__ D,                       // [NS][512], s-major
    const float* __restrict__ mean, const float* __restrict__ istd,
    const float* __restrict__ gamma, const float* __restrict__ beta,
    float* __restrict__ out) {                         // (b, 512, hw)
  __shared__ float t[32][33];
  const int b  = blockIdx.z;
  const int o0 = blockIdx.y * 32;
  const int s0 = blockIdx.x * 32;
  const int tc = threadIdx.x & 31, tr = threadIdx.x >> 5;
  const float* Dp = D + (long)b * HW * 512;
#pragma unroll
  for (int r = tr; r < 32; r += 8)
    t[r][tc] = Dp[(long)(s0 + r) * 512 + o0 + tc];     // coalesced in o
  __syncthreads();
  float* op = out + (long)b * 512 * HW;
#pragma unroll
  for (int r = tr; r < 32; r += 8) {
    int o = o0 + r;
    op[(long)o * HW + s0 + tc] =
        (t[tc][r] - mean[o]) * istd[o] * gamma[o] + beta[o];   // coalesced in s
  }
}

// --------- zero padded channels 537..543 of feat [s][544] ------------------
__global__ void zero_pad_feat(unsigned short* __restrict__ featb) {
  long i = (long)blockIdx.x * blockDim.x + threadIdx.x;
  if (i >= NS * 7) return;
  long s = i / 7;
  int  c = 537 + (int)(i - s * 7);
  featb[s * 544 + c] = 0;
}

// ----- 5x5 local self-correlation + L2 norm, NHWC, chans 512..536 ----------
#define CP 68   // padded per-pixel chunk stride (64 chans + pad)
__global__ __launch_bounds__(256) void local_corr(
    const unsigned short* __restrict__ feat, unsigned short* __restrict__ featw) {
  __shared__ unsigned short hs[400 * CP];     // 20x20 halo x 64-chan chunk
  const int b   = blockIdx.z;
  const int ty0 = blockIdx.y * 16, tx0 = blockIdx.x * 16;
  const int tx  = threadIdx.x & 15, ty = threadIdx.x >> 4;
  const unsigned short* base = feat + (long)b * HW * 544;

  float acc[25];
#pragma unroll
  for (int p = 0; p < 25; ++p) acc[p] = 0.f;

  for (int c0 = 0; c0 < 512; c0 += 64) {
    __syncthreads();
    for (int e = threadIdx.x; e < 400 * 16; e += 256) {   // uint2 granules
      int pix = e >> 4, q = e & 15;
      int r = pix / 20, col = pix - r * 20;
      int gy = ty0 + r - 2, gx = tx0 + col - 2;
      uint2 v = make_uint2(0u, 0u);
      if (gy >= 0 && gy < 64 && gx >= 0 && gx < 64)
        v = *(const uint2*)(base + (long)(gy * 64 + gx) * 544 + c0 + q * 4);
      *(uint2*)&hs[pix * CP + q * 4] = v;
    }
    __syncthreads();
    const unsigned short* ctr = &hs[((ty + 2) * 20 + (tx + 2)) * CP];
#pragma unroll 4
    for (int c = 0; c < 64; ++c) {
      float cv = bf2f(ctr[c]);
#pragma unroll
      for (int dy = 0; dy < 5; ++dy)
#pragma unroll
        for (int dx = 0; dx < 5; ++dx)
          acc[dy * 5 + dx] =
              fmaf(cv, bf2f(hs[((ty + dy) * 20 + (tx + dx)) * CP + c]), acc[dy * 5 + dx]);
    }
  }

  float ss = 1e-6f;
#pragma unroll
  for (int p = 0; p < 25; ++p) ss = fmaf(acc[p], acc[p], ss);
  float sc = rsqrtf(ss);

  unsigned short* o =
      featw + (long)b * HW * 544 + (long)((ty0 + ty) * 64 + tx0 + tx) * 544 + 512;
#pragma unroll
  for (int p = 0; p < 25; ++p) o[p] = f2bf(acc[p] * sc);
}

// ---------------------------------------------------------------------------
extern "C" void kernel_launch(void* const* d_in, const int* in_sizes, int n_in,
                              void* d_out, int out_size, void* d_ws, size_t ws_size,
                              hipStream_t stream) {
  const float* x     = (const float*)d_in[0];   // (16,1024,64,64)
  const float* w_in  = (const float*)d_in[1];   // (512,1024)
  const float* g1    = (const float*)d_in[2];
  const float* b1    = (const float*)d_in[3];
  const float* w_emb = (const float*)d_in[4];   // (512,537)
  const float* g2    = (const float*)d_in[5];
  const float* b2    = (const float*)d_in[6];
  const float* w_out = (const float*)d_in[7];   // (512,512)
  const float* g3    = (const float*)d_in[8];
  const float* b3    = (const float*)d_in[9];
  float* out = (float*)d_out;                   // (16,512,64,64)

  size_t off = 0;
  auto take = [&](size_t bytes) -> void* {
    void* p = (char*)d_ws + off;
    off += (bytes + 255) & ~(size_t)255;
    return p;
  };
  unsigned short* xb    = (unsigned short*)take((size_t)NS * 1024 * 2); // NHWC x
  unsigned short* winb  = (unsigned short*)take((size_t)512 * 1024 * 2);
  unsigned short* wembb = (unsigned short*)take((size_t)512 * 544 * 2);
  unsigned short* woutb = (unsigned short*)take((size_t)512 * 512 * 2);
  float*          bigf  = (float*)take((size_t)NS * 512 * 4);           // pre-BN, reused 3x
  unsigned short* featb = (unsigned short*)take((size_t)NS * 544 * 2);  // [h1|gs|pad] NHWC
  unsigned short* h2b   = (unsigned short*)take((size_t)NS * 512 * 2);
  float* psum = (float*)take(32 * 512 * 4);
  float* psq  = (float*)take(32 * 512 * 4);
  float* mean = (float*)take(512 * 4);
  float* istd = (float*)take(512 * 4);
  (void)ws_size; (void)in_sizes; (void)n_in; (void)out_size;

  // ---- layout conversion / packing ----
  transpose_pack_x<<<dim3(128, 32, NB), 256, 0, stream>>>(x, xb);
  pack_bf16<<<2048, 256, 0, stream>>>(w_in, winb, 512L * 1024);
  pack_wemb<<<(512 * 544 + 255) / 256, 256, 0, stream>>>(w_emb, wembb, 512, 537, 544);
  pack_bf16<<<1024, 256, 0, stream>>>(w_out, woutb, 512L * 512);

  const dim3 gblk(256);
  const dim3 ggrid(512 / TN, (unsigned)(NS / TM));   // (4, 512)

  // ---- stage 1: h1_pre = x_nhwc @ W_in^T ----
  gemm_nhwc_wmma<<<ggrid, gblk, 0, stream>>>(xb, winb, bigf, 1024);
  col_stats_partial<<<dim3(32, 2), 256, 0, stream>>>(bigf, psum, psq);
  col_stats_final<<<2, 256, 0, stream>>>(psum, psq, mean, istd);
  bn_relu_pack<<<8192, 256, 0, stream>>>(bigf, mean, istd, g1, b1, featb, 544);

  // ---- spatial descriptor ----
  zero_pad_feat<<<(unsigned)((NS * 7 + 255) / 256), 256, 0, stream>>>(featb);
  local_corr<<<dim3(4, 4, NB), 256, 0, stream>>>(featb, featb);

  // ---- stage 2: h2_pre = feat @ W_emb^T (K padded 537->544) ----
  gemm_nhwc_wmma<<<ggrid, gblk, 0, stream>>>(featb, wembb, bigf, 544);
  col_stats_partial<<<dim3(32, 2), 256, 0, stream>>>(bigf, psum, psq);
  col_stats_final<<<2, 256, 0, stream>>>(psum, psq, mean, istd);
  bn_relu_pack<<<8192, 256, 0, stream>>>(bigf, mean, istd, g2, b2, h2b, 512);

  // ---- stage 3: out_pre = h2 @ W_out^T ----
  gemm_nhwc_wmma<<<ggrid, gblk, 0, stream>>>(h2b, woutb, bigf, 512);
  col_stats_partial<<<dim3(32, 2), 256, 0, stream>>>(bigf, psum, psq);
  col_stats_final<<<2, 256, 0, stream>>>(psum, psq, mean, istd);
  bn_final_t<<<dim3(128, 16, NB), 256, 0, stream>>>(bigf, mean, istd, g3, b3, out);
}
